// LongformerSelfAttentionForMBart_25890062861075
// MI455X (gfx1250) — matrix operand order repaired and checked by reading, not compiled, not run on hardware
//
#include <hip/hip_runtime.h>
#include <hip/hip_bf16.h>

typedef __attribute__((ext_vector_type(16))) _Float16 v16h;
typedef __attribute__((ext_vector_type(8)))  _Float16 v8h;
typedef __attribute__((ext_vector_type(8)))  float    v8f;

#define S_LEN   4096
#define S_PAD   4608              /* S_LEN + 2*WWIN halo */
#define D_MODEL 768
#define N_HEADS 12
#define DH      64
#define WWIN    256
#define BLK     256
#define NBLK    16
#define NEGV    (-1.0e9f)

static __device__ __forceinline__ v8f wmma_f16(v16h a, v16h b, v8f c) {
  return __builtin_amdgcn_wmma_f32_16x16x32_f16(false, a, false, b, (short)0, c, false, false);
}

// ---- Fragment loaders (CDNA5 wave32 16x16x32 f16; this code uses contiguous k = 16*g + e in A and B) ----

// A 16x32 (MxK) from row-major source; lane m = lane&15, K striped:
// vgpr0..7 -> K = g*16 + {0..15}
// Loads are two contiguous 8-half (16B) groups per lane -> b128 friendly.
static __device__ __forceinline__ v16h load_afrag_rm(const _Float16* base, int ld,
                                                     int k0, int lane) {
  int m = lane & 15, g = lane >> 4;
  const _Float16* rp = base + (ptrdiff_t)m * ld + k0;
  v16h out;
#pragma unroll
  for (int i = 0; i < 8; ++i) {
    int k = g * 16 + 2 * i;
    out[2 * i]     = rp[k];
    out[2 * i + 1] = rp[k + 1];
  }
  return out;
}

// B 32x16 (KxN) where B[k][n] = rowmajor[n*ld + k] (i.e. B = Wt^T).
// Lane n = lane&15; K = 16*g + e  -> 16 contiguous halves (32B) per lane.
static __device__ __forceinline__ v16h load_bfragT(const _Float16* base, int ld,
                                                   int k0, int lane) {
  int n = lane & 15, g = lane >> 4;
  const _Float16* p = base + (ptrdiff_t)n * ld + k0 + g * 16;
  return *(const v16h*)p;
}

// ---- elementwise helpers ----
__global__ void cvt_f32_f16(const float* __restrict__ src, _Float16* __restrict__ dst, int n) {
  int i = blockIdx.x * blockDim.x + threadIdx.x;
  int stride = gridDim.x * blockDim.x;
  for (; i < n; i += stride) dst[i] = (_Float16)src[i];
}

__global__ void zero_f16(_Float16* __restrict__ dst, int n) {
  int i = blockIdx.x * blockDim.x + threadIdx.x;
  int stride = gridDim.x * blockDim.x;
  for (; i < n; i += stride) dst[i] = (_Float16)0.0f;
}

// ---- WMMA GEMM: C[m,n] = alpha * (A[m,:].Wt[n,:] + bias[n]) ----
// Wave tile: 32x32 (2x2 subtiles, 4 WMMAs per 32-wide k-step, 4 fragment
// loads per step -> 1.0 WMMA/fragment). K loop unrolled by 2 with disjoint
// ping-pong fragment sets (no rotate copies, ~96 live VGPRs -> no spills).
// mode 0: f16 row-major [M][N]                      (Q)
// mode 1: f16 V^T padded store [B][H][DH][S_PAD]    (V^T, +WWIN key halo)
// mode 2: f32 row-major [M][N]                      (final output)
// mode 3: f16 K padded-row store [B][S_PAD][N]      (K, +WWIN row halo)
__global__ void __launch_bounds__(256, 1)
gemm_wmma(const _Float16* __restrict__ A, const _Float16* __restrict__ Wt,
          const float* __restrict__ bias, void* __restrict__ outp,
          int M, int N, int K, float alpha, int mode) {
  int lane = threadIdx.x & 31;
  int wave = threadIdx.x >> 5;
  int tilesN2 = N / 32;
  int tileId = blockIdx.x * 8 + wave;
  if (tileId >= (M / 32) * tilesN2) return;
  int mt = tileId / tilesN2;
  int nt = tileId % tilesN2;
  int g = lane >> 4;

  v8f zero = {};
  v8f acc[2][2];
#pragma unroll
  for (int i = 0; i < 2; ++i)
#pragma unroll
    for (int j = 0; j < 2; ++j) acc[i][j] = zero;

  const _Float16* Arow0 = A + (size_t)mt * 32 * K;
  const _Float16* Arow1 = Arow0 + (size_t)16 * K;
  const _Float16* Wrow0 = Wt + (size_t)(nt * 32) * K;
  const _Float16* Wrow1 = Wrow0 + (size_t)16 * K;

  // Phase-0 prologue at k = 0.
  v16h a0[2], b0[2], a1[2], b1[2];
  a0[0] = load_afrag_rm(Arow0, K, 0, lane);
  a0[1] = load_afrag_rm(Arow1, K, 0, lane);
  b0[0] = load_bfragT(Wrow0, K, 0, lane);
  b0[1] = load_bfragT(Wrow1, K, 0, lane);

  for (int k0 = 0; k0 < K; k0 += 64) {   // K % 64 == 0
    // Load phase-1 fragments (k0+32) while phase-0 WMMAs run.
    a1[0] = load_afrag_rm(Arow0, K, k0 + 32, lane);
    a1[1] = load_afrag_rm(Arow1, K, k0 + 32, lane);
    b1[0] = load_bfragT(Wrow0, K, k0 + 32, lane);
    b1[1] = load_bfragT(Wrow1, K, k0 + 32, lane);
#pragma unroll
    for (int i = 0; i < 2; ++i)
#pragma unroll
      for (int j = 0; j < 2; ++j)
        acc[i][j] = wmma_f16(a0[i], b0[j], acc[i][j]);

    // Load next phase-0 fragments (k0+64) while phase-1 WMMAs run.
    int kn = k0 + 64;
    if (kn < K) {
      a0[0] = load_afrag_rm(Arow0, K, kn, lane);
      a0[1] = load_afrag_rm(Arow1, K, kn, lane);
      b0[0] = load_bfragT(Wrow0, K, kn, lane);
      b0[1] = load_bfragT(Wrow1, K, kn, lane);
    }
#pragma unroll
    for (int i = 0; i < 2; ++i)
#pragma unroll
      for (int j = 0; j < 2; ++j)
        acc[i][j] = wmma_f16(a1[i], b1[j], acc[i][j]);
  }

#pragma unroll
  for (int i = 0; i < 2; ++i) {
#pragma unroll
    for (int j = 0; j < 2; ++j) {
      int n = nt * 32 + j * 16 + (lane & 15);
      float bv = bias ? bias[n] : 0.0f;
#pragma unroll
      for (int r = 0; r < 8; ++r) {
        int m = mt * 32 + i * 16 + 8 * g + r;
        float val = alpha * (acc[i][j][r] + bv);
        if (mode == 0) {
          ((_Float16*)outp)[(size_t)m * N + n] = (_Float16)val;
        } else if (mode == 2) {
          ((float*)outp)[(size_t)m * N + n] = val;
        } else if (mode == 3) {          // K: padded rows
          int bb = m >> 12, s = m & (S_LEN - 1);
          ((_Float16*)outp)[((size_t)bb * S_PAD + s + WWIN) * N + n] = (_Float16)val;
        } else {                          // V^T: padded columns
          int bb = m >> 12, s = m & (S_LEN - 1);
          int h = n >> 6, d = n & (DH - 1);
          ((_Float16*)outp)[(((size_t)bb * N_HEADS + h) * DH + d) * S_PAD + s + WWIN] =
              (_Float16)val;
        }
      }
    }
  }
}

// ---- Sliding-window attention: online softmax, transposed-score trick ----
__global__ void __launch_bounds__(256, 1)
swattn_wmma(const _Float16* __restrict__ q16, const _Float16* __restrict__ kp16,
            const _Float16* __restrict__ vt16, const float* __restrict__ mask,
            _Float16* __restrict__ ctx16) {
  int lane = threadIdx.x & 31;
  int wave = threadIdx.x >> 5;
  int g = lane >> 4;
  int n = lane & 15;

  int id  = blockIdx.x;
  int blk = id % NBLK;
  int h   = (id / NBLK) % N_HEADS;
  int b   = id / (NBLK * N_HEADS);

  const _Float16* qbh  = q16 + ((size_t)b * S_LEN) * D_MODEL + h * DH;
  // Padded K: logical key s lives at row (s + WWIN); negative window keys stay in-bounds.
  const _Float16* kbh  = kp16 + ((size_t)b * S_PAD + WWIN) * D_MODEL + h * DH;
  // Padded V^T: logical key s at column (s + WWIN).
  const _Float16* vtbh = vt16 + (((size_t)b * N_HEADS + h) * DH) * S_PAD + WWIN;
  const float*    mb   = mask + (size_t)b * S_LEN;

  for (int qi = 0; qi < 2; ++qi) {
    int qt    = wave * 2 + qi;
    int qbase = blk * BLK + qt * 16;   // global query pos of tile row 0
    int qpos  = qbase + n;             // this lane's query (N column)

    // Q fragments (B operand of S^T = K_win * Q^T); reused for all key chunks.
    v16h Bq[2];
#pragma unroll
    for (int kk = 0; kk < 2; ++kk)
      Bq[kk] = load_bfragT(qbh + (size_t)qbase * D_MODEL, D_MODEL, kk * 32, lane);

    v8f zero = {};
    v8f acc[4];
#pragma unroll
    for (int d = 0; d < 4; ++d) acc[d] = zero;
    float m_run = -3.0e38f;
    float l_run = 0.0f;

    int kwin0 = blk * BLK - WWIN;
    for (int kc = 0; kc < 24; ++kc) {      // 768 window keys in chunks of 32
      int kstart = kwin0 + kc * 32;

      // S^T tiles: t0 = keys [kstart,+16), t1 = keys [kstart+16,+32).
      v8f t0 = zero, t1 = zero;
#pragma unroll
      for (int kk = 0; kk < 2; ++kk) {
        v16h a0 = load_afrag_rm(kbh + (ptrdiff_t)kstart * D_MODEL,        D_MODEL, kk * 32, lane);
        v16h a1 = load_afrag_rm(kbh + (ptrdiff_t)(kstart + 16) * D_MODEL, D_MODEL, kk * 32, lane);
        t0 = wmma_f16(a0, Bq[kk], t0);
        t1 = wmma_f16(a1, Bq[kk], t1);
      }

      // Issue V^T fragment loads early so they overlap the softmax VALU work.
      v16h av[4];
#pragma unroll
      for (int d = 0; d < 4; ++d)
        av[d] = load_afrag_rm(vtbh + (size_t)(16 * d) * S_PAD + kstart, S_PAD, 0, lane);

      // Prefetch next chunk's K rows / V^T lines (global_prefetch_b8).
      if (kc < 23) {
        int ksn = kstart + 32;
        const _Float16* pk = kbh + (ptrdiff_t)(ksn + lane) * D_MODEL;
        __builtin_prefetch(pk, 0, 1);
        __builtin_prefetch(pk + 32, 0, 1);
        const _Float16* pv = vtbh + (size_t)(2 * lane) * S_PAD + ksn;
        __builtin_prefetch(pv, 0, 1);
        __builtin_prefetch(pv + S_PAD, 0, 1);
      }

      // Band/bounds mask + key bias (-attention_mask), chunk row-max.
      float p0[8], p1[8];
      float lm = -3.0e38f;
#pragma unroll
      for (int r = 0; r < 8; ++r) {
        int key0 = kstart + 8 * g + r;
        int key1 = key0 + 16;
        int k0c = key0 < 0 ? 0 : (key0 >= S_LEN ? S_LEN - 1 : key0);
        int k1c = key1 < 0 ? 0 : (key1 >= S_LEN ? S_LEN - 1 : key1);
        float bias0 = mb[k0c], bias1 = mb[k1c];
        int rel0 = key0 - qpos, rel1 = key1 - qpos;
        bool v0 = (rel0 >= -WWIN) && (rel0 <= WWIN) && (key0 >= 0) && (key0 < S_LEN);
        bool v1 = (rel1 >= -WWIN) && (rel1 <= WWIN) && (key1 >= 0) && (key1 < S_LEN);
        p0[r] = v0 ? (t0[r] - bias0) : NEGV;
        p1[r] = v1 ? (t1[r] - bias1) : NEGV;
        lm = fmaxf(lm, fmaxf(p0[r], p1[r]));
      }
      lm = fmaxf(lm, __shfl_xor(lm, 16, 32));
      float nm = fmaxf(m_run, lm);
      float sc = __expf(m_run - nm);
      float rs = 0.0f;
#pragma unroll
      for (int r = 0; r < 8; ++r) {
        p0[r] = __expf(p0[r] - nm);
        p1[r] = __expf(p1[r] - nm);
        rs += p0[r] + p1[r];
      }
      rs += __shfl_xor(rs, 16, 32);
      l_run = l_run * sc + rs;
      m_run = nm;
#pragma unroll
      for (int d = 0; d < 4; ++d)
#pragma unroll
        for (int e = 0; e < 8; ++e) acc[d][e] *= sc;

      // Re-fragment P^T (C-layout) into B-layout for ctx^T = V^T * P^T:
      // B needs key = kstart + 16*g + e; own lane-group has 8 rows, partner
      // group (xor 16) supplies the other 8 via shuffles. No LDS needed.
      v16h pb;
#pragma unroll
      for (int r = 0; r < 8; ++r) {
        float s0 = __shfl_xor(p0[r], 16, 32);
        float s1 = __shfl_xor(p1[r], 16, 32);
        float lo = (g == 0) ? p0[r] : s1;   // e = r
        float hi = (g == 0) ? s0   : p1[r]; // e = 8 + r
        pb[r]     = (_Float16)lo;
        pb[8 + r] = (_Float16)hi;
      }

#pragma unroll
      for (int d = 0; d < 4; ++d)
        acc[d] = wmma_f16(av[d], pb, acc[d]);
    }

    // Normalize and store ctx row (token qpos): 8 contiguous halves per lane.
    float inv = 1.0f / l_run;
    size_t obase = ((size_t)b * S_LEN + qpos) * D_MODEL + h * DH;
#pragma unroll
    for (int d = 0; d < 4; ++d) {
      v8h pk;
#pragma unroll
      for (int r = 0; r < 8; ++r) pk[r] = (_Float16)(acc[d][r] * inv);
      *(v8h*)(ctx16 + obase + 16 * d + 8 * g) = pk;
    }
  }
}

extern "C" void kernel_launch(void* const* d_in, const int* in_sizes, int n_in,
                              void* d_out, int out_size, void* d_ws, size_t ws_size,
                              hipStream_t stream) {
  const float* hs = (const float*)d_in[0];
  const float* am = (const float*)d_in[1];
  const float* Wq = (const float*)d_in[2];
  const float* bq = (const float*)d_in[3];
  const float* Wk = (const float*)d_in[4];
  const float* bk = (const float*)d_in[5];
  const float* Wv = (const float*)d_in[6];
  const float* bv = (const float*)d_in[7];
  const float* Wo = (const float*)d_in[8];
  const float* bo = (const float*)d_in[9];

  const size_t MT = 2ull * S_LEN;   // 8192 tokens
  const size_t DM = D_MODEL;
  const size_t PADTOK = 2ull * S_PAD;   // padded token rows / key columns

  char* ws = (char*)d_ws;
  size_t off = 0;
  auto alloc = [&](size_t bytes) -> char* {
    char* p = ws + off;
    off += (bytes + 255) & ~(size_t)255;
    return p;
  };
  _Float16* hs16  = (_Float16*)alloc(MT * DM * 2);
  _Float16* wq16  = (_Float16*)alloc(DM * DM * 2);
  _Float16* wk16  = (_Float16*)alloc(DM * DM * 2);
  _Float16* wv16  = (_Float16*)alloc(DM * DM * 2);
  _Float16* wo16  = (_Float16*)alloc(DM * DM * 2);
  _Float16* q16   = (_Float16*)alloc(MT * DM * 2);
  _Float16* kp16  = (_Float16*)alloc(PADTOK * DM * 2);   // [B][S_PAD][768]
  _Float16* vt16  = (_Float16*)alloc(PADTOK * DM * 2);   // [B][H][64][S_PAD]
  _Float16* ctx16 = (_Float16*)alloc(MT * DM * 2);

  // Deterministic halo zeroing (harness does not re-poison between replays).
  zero_f16<<<512, 256, 0, stream>>>(kp16, (int)(PADTOK * DM));
  zero_f16<<<512, 256, 0, stream>>>(vt16, (int)(PADTOK * DM));

  cvt_f32_f16<<<512, 256, 0, stream>>>(hs, hs16, (int)(MT * DM));
  cvt_f32_f16<<<128, 256, 0, stream>>>(Wq, wq16, (int)(DM * DM));
  cvt_f32_f16<<<128, 256, 0, stream>>>(Wk, wk16, (int)(DM * DM));
  cvt_f32_f16<<<128, 256, 0, stream>>>(Wv, wv16, (int)(DM * DM));
  cvt_f32_f16<<<128, 256, 0, stream>>>(Wo, wo16, (int)(DM * DM));

  int tiles = (int)((MT / 32) * (DM / 32));   // 6144 wave tiles (32x32 each)
  int grid = (tiles + 7) / 8;                 // 768 workgroups (8 waves each)

  // Q pre-scaled by 1/sqrt(dh) = 1/8; K into padded rows; V^T into padded cols.
  gemm_wmma<<<grid, 256, 0, stream>>>(hs16, wq16, bq, q16,  (int)MT, (int)DM, (int)DM, 0.125f, 0);
  gemm_wmma<<<grid, 256, 0, stream>>>(hs16, wk16, bk, kp16, (int)MT, (int)DM, (int)DM, 1.0f, 3);
  gemm_wmma<<<grid, 256, 0, stream>>>(hs16, wv16, bv, vt16, (int)MT, (int)DM, (int)DM, 1.0f, 1);

  swattn_wmma<<<2 * N_HEADS * NBLK, 256, 0, stream>>>(q16, kp16, vt16, am, ctx16);

  gemm_wmma<<<grid, 256, 0, stream>>>(ctx16, wo16, bo, d_out, (int)MT, (int)DM, (int)DM, 1.0f, 2);
}
